// DenseEdgeConv_snn_NoisySAN_57664230916500
// MI455X (gfx1250) — matrix operand order repaired and checked by reading, not compiled, not run on hardware
//
#include <hip/hip_runtime.h>
#include <hip/hip_bf16.h>

// Problem constants (from reference): T=4, B=8, N=2048, d=g=32, K=16
#define TBF   32          // T*B flattened batch
#define NPTS  2048
#define DF    32          // feature dim d
#define KNN   16
#define WAVES 8           // waves per block (256 threads, wave32)
#define PTSW  4           // points processed per wave

typedef __attribute__((ext_vector_type(16))) __bf16    v16bf;
typedef __attribute__((ext_vector_type(8)))  float     v8f;
typedef __attribute__((ext_vector_type(4)))  unsigned  u32x4;   // trivial 16B vector

__device__ __forceinline__ unsigned packbf2(float a, float b) {
  union { __bf16 h[2]; unsigned u; } p;     // __bf16 is trivial -> union OK
  p.h[0] = (__bf16)a; p.h[1] = (__bf16)b;
  return p.u;                               // lowers to v_cvt_pk_bf16_f32
}
__device__ __forceinline__ unsigned short bfbits(float f) {
  union { __bf16 h; unsigned short s; } u; u.h = (__bf16)f; return u.s;
}

// A operand (16x32 bf16) from a row-major bf16 LDS tile.
// Lane holds row = lane&15; K pairs { (j<4 ? 2j : 2j+8) + 8*(lane>>4) } form
// two contiguous 16B runs -> 2x ds_load_b128.
__device__ __forceinline__ v16bf load_A(const unsigned short* base, int stride,
                                        int lane, int kbase) {
  const int r = lane & 15, half = lane >> 4;
  const unsigned short* p = base + r * stride + kbase + 8 * half;
  union { v16bf v; u32x4 q[2]; } a;
  a.q[0] = *(const u32x4*)(p);        // K = kbase+8h .. +7
  a.q[1] = *(const u32x4*)(p + 16);   // K = kbase+16+8h ..
  return a.v;
}

// B operand (32x16 bf16). B[k][col] = W[col][k], W row-major (out,cin) in LDS.
// Lane holds col = (lane&15)+16*ntile; K pairs { (j<4 ? 2j : 2j+8) + 8*(lane>>4) }
// form two contiguous 16B runs -> 2x ds_load_b128.
__device__ __forceinline__ v16bf load_B(const unsigned short* w, int cin,
                                        int lane, int ntile, int kbase) {
  const int col = (lane & 15) + (ntile << 4), half = lane >> 4;
  const unsigned short* p = w + col * cin + kbase + 8 * half;
  union { v16bf v; u32x4 q[2]; } b;
  b.q[0] = *(const u32x4*)(p);
  b.q[1] = *(const u32x4*)(p + 16);
  return b.v;
}

// One FC layer: 16 edge rows x (KCH*32) in -> 32 out, bf16 WMMA, f32 accum.
// Epilogue: bias (+ReLU), optional bf16 store into hbuf (row stride 128),
// and in-register max over the 16 edge rows written straight to global out.
template<int KCH, bool RELU, bool STORE>
__device__ __forceinline__ void fc_layer(const unsigned short* in, int instride,
                                         const unsigned short* w, const float* bias,
                                         unsigned short* outcols, float* omax,
                                         int lane) {
  v8f acc0 = {0.f, 0.f, 0.f, 0.f, 0.f, 0.f, 0.f, 0.f};
  v8f acc1 = acc0;
  const int cin = KCH * 32;
#pragma unroll
  for (int kc = 0; kc < KCH; ++kc) {
    v16bf A  = load_A(in, instride, lane, kc * 32);
    v16bf B0 = load_B(w, cin, lane, 0, kc * 32);
    acc0 = __builtin_amdgcn_wmma_f32_16x16x32_bf16(false, A, false, B0,
                                                   (short)0, acc0, false, false);
    v16bf B1 = load_B(w, cin, lane, 1, kc * 32);
    acc1 = __builtin_amdgcn_wmma_f32_16x16x32_bf16(false, A, false, B1,
                                                   (short)0, acc1, false, false);
  }
  // D layout: lane holds col = lane&15; VGPR v holds row v + 8*(lane>>4).
  const int col = lane & 15, half = lane >> 4;
  const float bb0 = bias[col], bb1 = bias[16 + col];
  float m0 = -3.0e38f, m1 = -3.0e38f;
#pragma unroll
  for (int v = 0; v < 8; ++v) {
    const int row = v + 8 * half;
    float y0 = acc0[v] + bb0;
    float y1 = acc1[v] + bb1;
    if (RELU) { y0 = fmaxf(y0, 0.f); y1 = fmaxf(y1, 0.f); }
    m0 = fmaxf(m0, y0); m1 = fmaxf(m1, y1);
    if (STORE) {
      outcols[row * 128 + col]      = bfbits(y0);
      outcols[row * 128 + 16 + col] = bfbits(y1);
    }
  }
  // combine the two row-halves across lane halves, then 32 lanes cover 32 cols
  m0 = fmaxf(m0, __shfl_xor(m0, 16, 32));
  m1 = fmaxf(m1, __shfl_xor(m1, 16, 32));
  omax[col + 16 * half] = half ? m1 : m0;
}

// ---------------- KNN: brute force top-17 per query point ----------------
__global__ __launch_bounds__(256) void knn_kernel(const float* __restrict__ pos,
                                                  int* __restrict__ idx) {
  __shared__ float sx[NPTS], sy[NPTS], sz[NPTS];
  const int bpb = NPTS / 256;                 // blocks per batch
  const int tb = blockIdx.x / bpb;
  const int q  = (blockIdx.x % bpb) * 256 + threadIdx.x;
  const float* p = pos + (size_t)tb * NPTS * 3;
  for (int i = threadIdx.x; i < NPTS; i += 256) {
    sx[i] = p[i * 3 + 0]; sy[i] = p[i * 3 + 1]; sz[i] = p[i * 3 + 2];
  }
  __syncthreads();
  const float qx = sx[q], qy = sy[q], qz = sz[q];
  float bd[KNN + 1]; int bi[KNN + 1];
#pragma unroll
  for (int t = 0; t <= KNN; ++t) { bd[t] = 3.0e38f; bi[t] = 0; }
  for (int c = 0; c < NPTS; ++c) {
    const float dx = sx[c] - qx, dy = sy[c] - qy, dz = sz[c] - qz;
    const float d = dx * dx + dy * dy + dz * dz;
    if (d < bd[KNN]) {
      bd[KNN] = d; bi[KNN] = c;
      // one bubble pass, static indices only (stays in VGPRs)
#pragma unroll
      for (int t = KNN; t > 0; --t) {
        if (bd[t] < bd[t - 1]) {
          const float td = bd[t]; bd[t] = bd[t - 1]; bd[t - 1] = td;
          const int   ti = bi[t]; bi[t] = bi[t - 1]; bi[t - 1] = ti;
        }
      }
    }
  }
  int* o = idx + ((size_t)tb * NPTS + q) * KNN;
#pragma unroll
  for (int t = 0; t < KNN; ++t) o[t] = bi[t + 1];  // skip self
}

// ---------------- EdgeConv MLP: one wave per point, WMMA bf16 ----------------
__global__ __launch_bounds__(256) void edgeconv_wmma(
    const float* __restrict__ x,
    const float* __restrict__ W0, const float* __restrict__ b0,
    const float* __restrict__ W1, const float* __restrict__ b1,
    const float* __restrict__ W2, const float* __restrict__ b2,
    const float* __restrict__ W3, const float* __restrict__ b3,
    const int* __restrict__ idx,
    float* __restrict__ out) {
  __shared__ __align__(16) unsigned short sW0[32 * 96];
  __shared__ __align__(16) unsigned short sW1[32 * 64];
  __shared__ __align__(16) unsigned short sW2[32 * 96];
  __shared__ __align__(16) unsigned short sW3[32 * 128];
  __shared__ float sB[128];
  __shared__ __align__(16) unsigned short ebuf[WAVES][16 * 96];   // [x | knn | knn-x]
  __shared__ __align__(16) unsigned short hbuf[WAVES][16 * 128];  // [y2 | y1 | y0 | x]

  const int tid = threadIdx.x;
  for (int i = tid; i < 32 * 96;  i += 256) sW0[i] = bfbits(W0[i]);
  for (int i = tid; i < 32 * 64;  i += 256) sW1[i] = bfbits(W1[i]);
  for (int i = tid; i < 32 * 96;  i += 256) sW2[i] = bfbits(W2[i]);
  for (int i = tid; i < 32 * 128; i += 256) sW3[i] = bfbits(W3[i]);
  if (tid < 32) {
    sB[tid] = b0[tid]; sB[32 + tid] = b1[tid];
    sB[64 + tid] = b2[tid]; sB[96 + tid] = b3[tid];
  }
  __syncthreads();

  const int wave = tid >> 5, lane = tid & 31;
  unsigned short* eb = ebuf[wave];
  unsigned short* hb = hbuf[wave];
  const int r = lane & 15, half = lane >> 4, c0 = half << 4;

  for (int it = 0; it < PTSW; ++it) {
    const int point = blockIdx.x * (WAVES * PTSW) + it * WAVES + wave;
    const int tb = point >> 11;
    const float* xc = x + (size_t)point * DF;

    if (it + 1 < PTSW)
      __builtin_prefetch(idx + (size_t)(point + WAVES) * KNN, 0, 0);

    // Row r's neighbor; lanes 0-15 handle cols 0..15, lanes 16-31 cols 16..31.
    const int m = idx[(size_t)point * KNN + r];
    const float* xn = x + (size_t)(tb * NPTS + m) * DF;

    float cx[16], nx[16];
#pragma unroll
    for (int j = 0; j < 16; ++j) cx[j] = xc[c0 + j];
#pragma unroll
    for (int j = 0; j < 16; ++j) nx[j] = xn[c0 + j];

    // Build edge features; 16 contiguous bf16 per region -> 2x b128 stores.
#pragma unroll
    for (int g = 0; g < 2; ++g) {
      u32x4 qx, qn, qd;
#pragma unroll
      for (int j = 0; j < 4; ++j) {
        const int e = 8 * g + 2 * j;
        qx[j] = packbf2(cx[e], cx[e + 1]);
        qn[j] = packbf2(nx[e], nx[e + 1]);
        qd[j] = packbf2(nx[e] - cx[e], nx[e + 1] - cx[e + 1]);
      }
      *(u32x4*)(eb + r * 96 + c0 + 8 * g)        = qx;  // x
      *(u32x4*)(eb + r * 96 + 32 + c0 + 8 * g)   = qn;  // knn
      *(u32x4*)(eb + r * 96 + 64 + c0 + 8 * g)   = qd;  // knn - x
      *(u32x4*)(hb + r * 128 + 96 + c0 + 8 * g)  = qx;  // x block of h
    }
    __builtin_amdgcn_wave_barrier();

    // Output channel map: [y3(0:32), y2(32:64), y1(64:96), y0(96:128), x(128:160)]
    float* op = out + (size_t)point * 160;

    fc_layer<3, true,  true >(eb,       96, sW0, sB,      hb + 64, op + 96, lane);
    __builtin_amdgcn_wave_barrier();
    fc_layer<2, true,  true >(hb + 64, 128, sW1, sB + 32, hb + 32, op + 64, lane);
    __builtin_amdgcn_wave_barrier();
    fc_layer<3, true,  true >(hb + 32, 128, sW2, sB + 64, hb + 0,  op + 32, lane);
    __builtin_amdgcn_wave_barrier();
    fc_layer<4, false, false>(hb + 0,  128, sW3, sB + 96, hb,      op + 0,  lane);

    op[128 + lane] = xc[lane];  // x_rep is constant over K -> max == x
  }
}

extern "C" void kernel_launch(void* const* d_in, const int* in_sizes, int n_in,
                              void* d_out, int out_size, void* d_ws, size_t ws_size,
                              hipStream_t stream) {
  const float* x   = (const float*)d_in[0];
  const float* pos = (const float*)d_in[1];
  const float* W0  = (const float*)d_in[2]; const float* b0 = (const float*)d_in[3];
  const float* W1  = (const float*)d_in[4]; const float* b1 = (const float*)d_in[5];
  const float* W2  = (const float*)d_in[6]; const float* b2 = (const float*)d_in[7];
  const float* W3  = (const float*)d_in[8]; const float* b3 = (const float*)d_in[9];
  float* out = (float*)d_out;
  int* idx = (int*)d_ws;  // TBF*NPTS*KNN ints = 4 MB of scratch

  knn_kernel<<<TBF * (NPTS / 256), 256, 0, stream>>>(pos, idx);
  edgeconv_wmma<<<(TBF * NPTS) / (WAVES * PTSW), 256, 0, stream>>>(
      x, W0, b0, W1, b1, W2, b2, W3, b3, idx, out);
}
